// LSTMNet_60894046323378
// MI455X (gfx1250) — matrix operand order repaired and checked by reading, not compiled, not run on hardware
//
#include <hip/hip_runtime.h>

// Problem constants (match reference): B=128, S=512, I=256, H=512
#define B_ 128
#define S_ 512
#define I_ 256
#define H_ 512

typedef __attribute__((ext_vector_type(16))) __bf16 v16bf;
typedef __attribute__((ext_vector_type(8)))  float  v8f;
typedef __attribute__((ext_vector_type(4)))  unsigned int u32x4;

// explicit global-address-space types: keep weight loads on the
// global_load_b128 path (LOADcnt only) even through the anti-LICM asm barrier
typedef __attribute__((address_space(1))) const unsigned short gus_t;
typedef __attribute__((address_space(1))) const u32x4         gu32x4_t;

#if __has_builtin(__builtin_amdgcn_sched_barrier)
#define SCHED_FENCE() __builtin_amdgcn_sched_barrier(0)
#else
#define SCHED_FENCE()
#endif

__device__ __forceinline__ unsigned short f2bf(float f) {
    unsigned u = __float_as_uint(f);
    unsigned r = u + 0x7FFFu + ((u >> 16) & 1u);   // round-to-nearest-even
    return (unsigned short)(r >> 16);
}

__device__ __forceinline__ float bf2f(unsigned short b) {
    return __uint_as_float(((unsigned)b) << 16);
}

__device__ __forceinline__ float sigf(float x)  { return 1.0f / (1.0f + __expf(-x)); }
__device__ __forceinline__ float tanhf_(float x){ return 1.0f - 2.0f / (__expf(2.0f * x) + 1.0f); }

__device__ __forceinline__ v8f splat8(float x) {
    v8f v;
#pragma unroll
    for (int i = 0; i < 8; ++i) v[i] = x;
    return v;
}

// A-fragment (16x32 bf16, MxK), ISA layout => 2x ds_load_b128 per lane
__device__ __forceinline__ v16bf load_a_frag(const unsigned short* base, int ld,
                                             int M, int kbase, int hiHalf) {
    union { v16bf v; unsigned u[8]; } f;
    const unsigned short* p = base + M * ld + kbase + (hiHalf ? 8 : 0);
#pragma unroll
    for (int q = 0; q < 8; ++q) {
        int k = 2 * q + (q >= 4 ? 8 : 0);
        f.u[q] = *(const unsigned*)(p + k);
    }
    return f.v;
}

// B-fragment: lane L reads 16 contiguous bf16 (32B) = 2x global_load_b128
__device__ __forceinline__ v16bf load_b_frag(gus_t* p) {
    union { v16bf v; u32x4 q[2]; } f;
    gu32x4_t* g = (gu32x4_t*)p;
    f.q[0] = g[0];
    f.q[1] = g[1];
    return f.v;
}

// Pack W (row-major [4H, K]) into per-tile B-fragment layout:
// tile (tn, tk): N = tn*16.., K = tk*32.. ; lane L, e=0..15:
//   N=L&15, h=L>>4: K = e<8 ? 8h+e : 16+8h+(e-8)
__global__ __launch_bounds__(256) void pack_weights_kernel(
        const float* __restrict__ Wih, const float* __restrict__ Whh,
        unsigned short* __restrict__ pih, unsigned short* __restrict__ phh) {
    const int NIH = 4 * H_ * I_;   // 524288
    const int NHH = 4 * H_ * H_;   // 1048576
    int idx = blockIdx.x * blockDim.x + threadIdx.x;
    if (idx < NIH) {
        int tile = idx >> 9, r = idx & 511;
        int L = r >> 4, e = r & 15;
        int tn = tile >> 3, tk = tile & 7;                 // KT = 8
        int n = tn * 16 + (L & 15);
        int k = tk * 32 + ((e < 8) ? 8 * (L >> 4) + e : 16 + 8 * (L >> 4) + (e - 8));
        pih[idx] = f2bf(Wih[n * I_ + k]);
    } else if (idx < NIH + NHH) {
        int j = idx - NIH;
        int tile = j >> 9, r = j & 511;
        int L = r >> 4, e = r & 15;
        int tn = tile >> 4, tk = tile & 15;                // KT = 16
        int n = tn * 16 + (L & 15);
        int k = tk * 32 + ((e < 8) ? 8 * (L >> 4) + e : 16 + 8 * (L >> 4) + (e - 8));
        phh[j] = f2bf(Whh[n * H_ + k]);
    }
}

// Persistent LSTM: grid = 16 WGs (2 sequences x 8 batch-tiles of 16 rows),
// block = 512 threads = 16 waves. Wave w owns hidden columns [w*32, w*32+32).
// c in fp32 VGPRs; h double-buffered in LDS (bf16) -> 1 barrier per timestep.
// K=768 GEMM is software-pipelined: 48 half-groups of 4 fragments,
// double-buffered, refills issued one full group ahead of use.
__global__ __launch_bounds__(512, 1) void lstm_persistent(
        const float* __restrict__ x1, const float* __restrict__ x2,
        const unsigned short* __restrict__ pih_in, const unsigned short* __restrict__ phh_in,
        const float* __restrict__ b_ih, const float* __restrict__ b_hh,
        float* __restrict__ out) {
    __shared__ unsigned short xs[2][16][I_];   // x_t tiles, bf16 (16 KB)
    __shared__ unsigned short hs[2][16][H_];   // h tiles, bf16 (32 KB)

    const int tid    = threadIdx.x;
    const int lane   = tid & 31;
    const int hiHalf = (lane >= 16);
    const int Mrow   = lane & 15;
    // wave id: force into SGPR so all tile bases are scalar
    const int wvu = __builtin_amdgcn_readfirstlane(tid >> 5);   // 0..15

    const int seq = blockIdx.x >> 3;      // 0 = text_one, 1 = text_two
    const int bt  = blockIdx.x & 7;       // batch tile (16 rows)
    const float* xbase = ((seq == 0) ? x1 : x2) + (size_t)(bt * 16) * S_ * I_;

    // prologue: zero h0 buffer, stage x(0) into xs[0]
    {
        unsigned* hz = (unsigned*)&hs[0][0][0];
        for (int i = tid; i < (16 * H_) / 2; i += 512) hz[i] = 0u;
        for (int i = tid; i < 16 * I_; i += 512) {
            int r = i >> 8, k = i & 255;
            xs[0][r][k] = f2bf(xbase[(size_t)r * S_ * I_ + k]);
        }
    }

    // biases (constant over time): col = g*H + wv*32 + u*16 + (lane&15)
    float bias[4][2];
#pragma unroll
    for (int g = 0; g < 4; ++g)
#pragma unroll
        for (int u = 0; u < 2; ++u) {
            int col = g * H_ + wvu * 32 + u * 16 + Mrow;
            bias[g][u] = b_ih[col] + b_hh[col];
        }

    v8f c[2];
    c[0] = splat8(0.0f);
    c[1] = splat8(0.0f);

    gus_t* pih = (gus_t*)pih_in;
    gus_t* phh = (gus_t*)phh_in;
    const int laneOff = lane * 16;        // bf16 elements (32 B per lane)

    // fragment address for k-tile tk (0..23) and tile jj (0..7 = g*2+u)
#define FRAG_PTR(tk, jj)                                                        \
    (((tk) < 8)                                                                 \
         ? (pih + ((((jj) >> 1) * 32 + wvu * 2 + ((jj) & 1)) * 8 + (tk)) * 512  \
                + laneOff)                                                      \
         : (phh + ((((jj) >> 1) * 32 + wvu * 2 + ((jj) & 1)) * 16 + (tk) - 8) * 512 \
                + laneOff))

    // A-fragment for k-tile tk from the current LDS buffers
#define A_FRAG(tk, cur)                                                         \
    (((tk) < 8) ? load_a_frag(&xs[(cur)][0][0], I_, Mrow, (tk) * 32, hiHalf)    \
                : load_a_frag(&hs[(cur)][0][0], H_, Mrow, ((tk) - 8) * 32, hiHalf))

    for (int t = 0; t < S_; ++t) {
        const int cur = t & 1, nxt = cur ^ 1;

        // defeat LICM each iteration (keeps weights streaming from L2, no spill)
        asm volatile("" : "+s"(pih), "+s"(phh));

        // pre-barrier: warm x(t+1) (zero-register prefetch) and issue the first
        // half-group of weight loads so the barrier wait hides their latency
        if (t + 1 < S_) {
#pragma unroll
            for (int i = 0; i < 8; ++i) {
                int idx = tid + i * 512;
                int r = idx >> 8, k = idx & 255;
                __builtin_prefetch(xbase + (size_t)r * S_ * I_ + (size_t)(t + 1) * I_ + k, 0, 3);
            }
        }
        v16bf bufA[4], bufB[4], afr[2];
#pragma unroll
        for (int j = 0; j < 4; ++j) bufA[j] = load_b_frag(FRAG_PTR(0, j));

        __syncthreads();  // buffers for step t (hs[cur], xs[cur]) fully written

        afr[0] = A_FRAG(0, cur);
#pragma unroll
        for (int j = 0; j < 4; ++j) bufB[j] = load_b_frag(FRAG_PTR(0, 4 + j));

        // accumulator init: re-materialize the bias splats every step (8 v_movs
        // each) instead of letting LICM keep 8 loop-invariant v8f splats (64
        // VGPRs) live across the time loop -> avoids the scratch spill/reload
        float bl[4][2];
#pragma unroll
        for (int g = 0; g < 4; ++g)
#pragma unroll
            for (int u = 0; u < 2; ++u) bl[g][u] = bias[g][u];
        asm volatile("" : "+v"(bl[0][0]), "+v"(bl[0][1]), "+v"(bl[1][0]),
                          "+v"(bl[1][1]), "+v"(bl[2][0]), "+v"(bl[2][1]),
                          "+v"(bl[3][0]), "+v"(bl[3][1]));
        v8f acc[4][2];
#pragma unroll
        for (int g = 0; g < 4; ++g)
#pragma unroll
            for (int u = 0; u < 2; ++u) acc[g][u] = splat8(bl[g][u]);

        // 48 half-groups: hg = 2*tk + half ; consume buf(hg), refill for hg+2
#pragma unroll
        for (int hg = 0; hg < 48; ++hg) {
            const int tk = hg >> 1, half = hg & 1;
#pragma unroll
            for (int j = 0; j < 4; ++j) {
                const int jj = half * 4 + j, g = jj >> 1, u = jj & 1;
                acc[g][u] = __builtin_amdgcn_wmma_f32_16x16x32_bf16(
                    false, afr[tk & 1], false, (half ? bufB : bufA)[j],
                    (short)0, acc[g][u], false, false);
            }
            SCHED_FENCE();
            if (half == 0 && tk + 1 < 24)            // A for next k-tile (ds)
                afr[(tk + 1) & 1] = A_FRAG(tk + 1, cur);
            if (hg + 2 < 48) {                       // refill consumed buffer
#pragma unroll
                for (int j = 0; j < 4; ++j) {
                    if (half)
                        bufB[j] = load_b_frag(FRAG_PTR(tk + 1, 4 + j));
                    else
                        bufA[j] = load_b_frag(FRAG_PTR(tk + 1, j));
                }
            }
            SCHED_FENCE();
        }

        // gates + state update; write new h into the OTHER buffer (no 2nd barrier)
#pragma unroll
        for (int u = 0; u < 2; ++u) {
#pragma unroll
            for (int r = 0; r < 8; ++r) {
                float iv = sigf(acc[0][u][r]);
                float fv = sigf(acc[1][u][r]);
                float gv = tanhf_(acc[2][u][r]);
                float ov = sigf(acc[3][u][r]);
                float cv = fv * c[u][r] + iv * gv;
                c[u][r] = cv;
                float hv = ov * tanhf_(cv);
                int mr = r + (hiHalf ? 8 : 0);
                hs[nxt][mr][wvu * 32 + u * 16 + Mrow] = f2bf(hv);
            }
        }

        // stage x(t+1) (prefetched above -> near-cache hit) into the other buffer
        if (t + 1 < S_) {
#pragma unroll
            for (int i = 0; i < 8; ++i) {
                int idx = tid + i * 512;
                int r = idx >> 8, k = idx & 255;
                xs[nxt][r][k] = f2bf(xbase[(size_t)r * S_ * I_ + (size_t)(t + 1) * I_ + k]);
            }
        }
        // next iteration's barrier separates these writes from the next reads
    }
    __syncthreads();

    // final h lives in buffer (S_ & 1) = 0 ; emit bf16 -> f32
    float* ob = out + (size_t)seq * B_ * H_ + (size_t)(bt * 16) * H_;
    for (int i = tid; i < 16 * H_; i += 512) {
        int r = i >> 9, n = i & 511;
        ob[(size_t)r * H_ + n] = bf2f(hs[0][r][n]);
    }
#undef FRAG_PTR
#undef A_FRAG
}

extern "C" void kernel_launch(void* const* d_in, const int* in_sizes, int n_in,
                              void* d_out, int out_size, void* d_ws, size_t ws_size,
                              hipStream_t stream) {
    (void)in_sizes; (void)n_in; (void)out_size; (void)ws_size;
    const float* x1  = (const float*)d_in[0];
    const float* x2  = (const float*)d_in[1];
    const float* Wih = (const float*)d_in[2];
    const float* Whh = (const float*)d_in[3];
    const float* bih = (const float*)d_in[4];
    const float* bhh = (const float*)d_in[5];

    unsigned short* pih = (unsigned short*)d_ws;          // 1 MB
    unsigned short* phh = pih + 4 * H_ * I_;              // 2 MB

    const int total = 4 * H_ * I_ + 4 * H_ * H_;          // 1,572,864 packed bf16
    pack_weights_kernel<<<(total + 255) / 256, 256, 0, stream>>>(Wih, Whh, pih, phh);

    lstm_persistent<<<16, 512, 0, stream>>>(x1, x2, pih, phh, bih, bhh, (float*)d_out);
}
